// GeneratorBlock_21809843929329
// MI455X (gfx1250) — matrix operand order repaired and checked by reading, not compiled, not run on hardware
//
#include <hip/hip_runtime.h>

typedef __attribute__((ext_vector_type(16))) __bf16 v16bf;
typedef __attribute__((ext_vector_type(8)))  __bf16 v8bf;
typedef __attribute__((ext_vector_type(8)))  float  v8f;
typedef __attribute__((ext_vector_type(4)))  int    v4i;

#define BATCH 16
#define CIN   512
#define COUT  512
#define WDIM  512
#define H     64
#define W     64
#define HW    (H*W)

// workspace byte offsets
#define WS_S     0
#define WS_DEMOD (WS_S + BATCH*CIN*4)
#define WS_W2    (WS_DEMOD + BATCH*COUT*4)
#define WS_WBF   (WS_W2 + COUT*CIN*4)

// gfx1250 async global->LDS copy (guarded: falls back to plain b128 copies)
#if defined(__HIP_DEVICE_COMPILE__) && __has_builtin(__builtin_amdgcn_global_load_async_to_lds_b128)
#define USE_ASYNC_LDS 1
typedef __attribute__((address_space(1))) v4i as1_v4i;
typedef __attribute__((address_space(3))) v4i as3_v4i;
#else
#define USE_ASYNC_LDS 0
#endif

__device__ __forceinline__ void wait_async_all() {
#if defined(__HIP_DEVICE_COMPILE__)
#if __has_builtin(__builtin_amdgcn_s_wait_asynccnt)
  __builtin_amdgcn_s_wait_asynccnt(0);
#elif USE_ASYNC_LDS
  asm volatile("s_wait_asynccnt 0" ::: "memory");
#endif
#endif
}

// ---------------- setup kernels ----------------

__global__ void style_kernel(const float* __restrict__ w_embs,
                             const float* __restrict__ style_w,
                             const float* __restrict__ style_b,
                             float* __restrict__ s) {
  int idx = blockIdx.x * blockDim.x + threadIdx.x;
  if (idx >= BATCH * CIN) return;
  int b = idx / CIN, i = idx % CIN;
  const float* we = w_embs + b * WDIM;
  const float* sw = style_w + i * WDIM;
  float acc = 0.f;
  for (int j = 0; j < WDIM; ++j) acc += we[j] * sw[j];
  s[idx] = acc + style_b[i];
}

__global__ void w2_kernel(const float* __restrict__ weight, float* __restrict__ w2) {
  int idx = blockIdx.x * blockDim.x + threadIdx.x;
  if (idx >= COUT * CIN) return;
  const float* wp = weight + (size_t)idx * 9;
  float a = 0.f;
  #pragma unroll
  for (int k = 0; k < 9; ++k) a += wp[k] * wp[k];
  w2[idx] = a;
}

__global__ void demod_kernel(const float* __restrict__ w2,
                             const float* __restrict__ s,
                             float* __restrict__ demod) {
  int idx = blockIdx.x * blockDim.x + threadIdx.x;
  if (idx >= BATCH * COUT) return;
  int b = idx / COUT, o = idx % COUT;
  const float* w2o = w2 + o * CIN;
  const float* sb  = s + b * CIN;
  float acc = 0.f;
  for (int i = 0; i < CIN; ++i) { float sv = sb[i]; acc += w2o[i] * sv * sv; }
  float norm = sqrtf(acc);
  demod[idx] = 1.f / fmaxf(norm, 1e-12f);
}

// repack weight [cout,cin,3,3] f32 -> wbf [cc=cin/32][cout][tap 9][ci 32] bf16
__global__ void wcvt_kernel(const float* __restrict__ weight, __bf16* __restrict__ wbf) {
  int idx = blockIdx.x * blockDim.x + threadIdx.x;
  if (idx >= COUT * CIN * 9) return;
  int o   = idx / (CIN * 9);
  int rem = idx - o * (CIN * 9);
  int i   = rem / 9;
  int t   = rem - i * 9;
  int cc  = i >> 5, ci = i & 31;
  size_t dst = (((size_t)cc * COUT + o) * 9 + t) * 32 + ci;
  wbf[dst] = (__bf16)weight[idx];
}

// ---------------- main implicit-GEMM conv via bf16 WMMA ----------------

#define SA 104  // lA row stride in halfs (96 data + 8 pad) -> 208B, 16B aligned
#define SB 40   // lB row stride in halfs (32 data + 8 pad) -> 80B,  16B aligned

__launch_bounds__(256)
__global__ void conv_kernel(const float* __restrict__ imgs,
                            const float* __restrict__ noise,
                            const float* __restrict__ bias,
                            const float* __restrict__ noise_w,
                            const float* __restrict__ s,
                            const float* __restrict__ demod,
                            const __bf16* __restrict__ wbf,
                            float* __restrict__ out) {
  __shared__ __attribute__((aligned(16))) __bf16 lA[128 * SA]; // [cout 128][3 kw * 32 ci]
  __shared__ __attribute__((aligned(16))) __bf16 lB[66 * SB];  // [padded col 66][ci 32]

  const int tid = threadIdx.x;
  const int y   = blockIdx.x;        // output row
  const int o0  = blockIdx.y * 128;  // cout tile base
  const int b   = blockIdx.z;

  const int wave = tid >> 5;
  const int lane = tid & 31;
  const int half = lane >> 4;
  const int lrow = lane & 15;
  const int wm   = wave >> 1;   // 0..3 -> 32-row M sub-tile
  const int wn   = wave & 1;    // 0..1 -> 32-col N sub-tile

  // B staging mapping: lanes sweep x (coalesced), groups of 8 ci per thread
  const int bc = tid & 63;      // source column 0..63 -> LDS col bc+1
  const int bg = tid >> 6;      // ci group 0..3

  // A staging mapping: 2 threads per cout row, 96B each (6 x b128)
  const int ar = tid >> 1;
  const int ap = tid & 1;

  v8f acc[2][2] = {};
  union Frag { v16bf v; v8bf h[2]; };

  for (int cc = 0; cc < CIN / 32; ++cc) {
    const int ci0 = cc * 32;

    float sv[8];
    #pragma unroll
    for (int j = 0; j < 8; ++j) sv[j] = s[b * CIN + ci0 + bg * 8 + j];

    const __bf16* aRow = wbf + (((size_t)cc * COUT + (o0 + ar)) * 9) * 32;

    for (int kh = 0; kh < 3; ++kh) {
      const int ysrc = y + kh - 1;
      const bool rowOK = ((unsigned)ysrc < (unsigned)H);

      __syncthreads();  // previous stage's compute done before overwrite

      // ---- stage A: 128 x 96 bf16 (3 kw taps), contiguous copies ----
      {
        const uint4* sp = (const uint4*)(aRow + (kh * 3) * 32 + ap * 48);
        uint4* dp = (uint4*)&lA[ar * SA + ap * 48];
#if USE_ASYNC_LDS
        #pragma unroll
        for (int k = 0; k < 6; ++k)
          __builtin_amdgcn_global_load_async_to_lds_b128(
              (as1_v4i*)(unsigned long long)(const void*)(sp + k),
              (as3_v4i*)(dp + k), 0, 0);
#else
        #pragma unroll
        for (int k = 0; k < 6; ++k) dp[k] = sp[k];
#endif
        // prefetch next cin-chunk's weights (gfx1250 global_prefetch)
        __builtin_prefetch(
            wbf + ((((size_t)((cc + 1) & (CIN / 32 - 1)) * COUT + (o0 + ar)) * 9) + kh * 3) * 32,
            0, 0);
      }

      // ---- stage B: padded row [66][32], modulated + bf16, branch-free ----
      {
        v8bf pk;
        if (rowOK) {
          const float* ip =
              imgs + ((size_t)(b * CIN + ci0 + bg * 8) * H + ysrc) * W + bc;
          #pragma unroll
          for (int j = 0; j < 8; ++j)
            pk[j] = (__bf16)(ip[(size_t)j * HW] * sv[j]);
        } else {
          #pragma unroll
          for (int j = 0; j < 8; ++j) pk[j] = (__bf16)0.f;
        }
        *(v8bf*)&lB[(bc + 1) * SB + bg * 8] = pk;
        if (bc == 0) {  // zero the halo columns
          v8bf z;
          #pragma unroll
          for (int j = 0; j < 8; ++j) z[j] = (__bf16)0.f;
          *(v8bf*)&lB[0 * SB + bg * 8]  = z;
          *(v8bf*)&lB[65 * SB + bg * 8] = z;
        }
      }

#if USE_ASYNC_LDS
      wait_async_all();
#endif
      __syncthreads();

      // ---- compute: 3 kw taps x (2x2) WMMA per wave ----
      #pragma unroll
      for (int kw = 0; kw < 3; ++kw) {
        Frag aF[2], bF[2];
        #pragma unroll
        for (int mi = 0; mi < 2; ++mi) {
          const int base = (wm * 32 + mi * 16 + lrow) * SA + kw * 32;
          aF[mi].h[0] = *(const v8bf*)&lA[base + half * 8];       // K 0..7 / 8..15
          aF[mi].h[1] = *(const v8bf*)&lA[base + 16 + half * 8];  // K 16..23 / 24..31
        }
        #pragma unroll
        for (int ni = 0; ni < 2; ++ni) {
          const int col = wn * 32 + ni * 16 + lrow + kw;          // shifted halo col
          bF[ni].h[0] = *(const v8bf*)&lB[col * SB + half * 8];
          bF[ni].h[1] = *(const v8bf*)&lB[col * SB + 16 + half * 8];
        }
        #pragma unroll
        for (int mi = 0; mi < 2; ++mi)
          #pragma unroll
          for (int ni = 0; ni < 2; ++ni)
            acc[mi][ni] = __builtin_amdgcn_wmma_f32_16x16x32_bf16(
                false, aF[mi].v, false, bF[ni].v, (short)0, acc[mi][ni], false, false);
      }
    }
  }

  // ---- epilogue: demod, bias, noise, LeakyReLU(0.2) ----
  #pragma unroll
  for (int mi = 0; mi < 2; ++mi) {
    float dmd[8], bs[8], nwv[8];
    #pragma unroll
    for (int v = 0; v < 8; ++v) {
      const int o = o0 + wm * 32 + mi * 16 + half * 8 + v;
      dmd[v] = demod[b * COUT + o];
      bs[v]  = bias[o];
      nwv[v] = noise_w[o];
    }
    #pragma unroll
    for (int ni = 0; ni < 2; ++ni) {
      const int x = wn * 32 + ni * 16 + lrow;
      const float nz = noise[b * HW + y * W + x];
      #pragma unroll
      for (int v = 0; v < 8; ++v) {
        const int o = o0 + wm * 32 + mi * 16 + half * 8 + v;
        float val = acc[mi][ni][v] * dmd[v] + bs[v] + nz * nwv[v];
        val = (val >= 0.f) ? val : 0.2f * val;
        out[((size_t)(b * COUT + o) * H + y) * W + x] = val;
      }
    }
  }
}

// ---------------- launcher ----------------

extern "C" void kernel_launch(void* const* d_in, const int* in_sizes, int n_in,
                              void* d_out, int out_size, void* d_ws, size_t ws_size,
                              hipStream_t stream) {
  const float* imgs    = (const float*)d_in[0];
  const float* w_embs  = (const float*)d_in[1];
  const float* noise   = (const float*)d_in[2];
  const float* weight  = (const float*)d_in[3];
  const float* bias    = (const float*)d_in[4];
  const float* noise_w = (const float*)d_in[5];
  const float* style_w = (const float*)d_in[6];
  const float* style_b = (const float*)d_in[7];
  float* out = (float*)d_out;

  char* ws = (char*)d_ws;
  float*  s     = (float*)(ws + WS_S);
  float*  demod = (float*)(ws + WS_DEMOD);
  float*  w2    = (float*)(ws + WS_W2);
  __bf16* wbf   = (__bf16*)(ws + WS_WBF);

  style_kernel<<<(BATCH * CIN + 255) / 256, 256, 0, stream>>>(w_embs, style_w, style_b, s);
  w2_kernel<<<(COUT * CIN + 255) / 256, 256, 0, stream>>>(weight, w2);
  demod_kernel<<<(BATCH * COUT + 255) / 256, 256, 0, stream>>>(w2, s, demod);
  wcvt_kernel<<<(9 * COUT * CIN + 255) / 256, 256, 0, stream>>>(weight, wbf);

  conv_kernel<<<dim3(H, COUT / 128, BATCH), 256, 0, stream>>>(
      imgs, noise, bias, noise_w, s, demod, wbf, out);
}
